// SelfAttention_47871705481474
// MI455X (gfx1250) — matrix operand order repaired and checked by reading, not compiled, not run on hardware
//
#include <hip/hip_runtime.h>

#define SEQ   2048
#define DM    768
#define NH    12
#define DH    64
#define BATCH 2

typedef __attribute__((ext_vector_type(16))) _Float16 v16h;
typedef __attribute__((ext_vector_type(4)))  _Float16 v4h;
typedef __attribute__((ext_vector_type(8)))  float    v8f;

__device__ __forceinline__ v8f wmma_f16(v16h a, v16h b, v8f c) {
  // D = A(16x32 f16) * B(32x16 f16) + C(16x16 f32)
  return __builtin_amdgcn_wmma_f32_16x16x32_f16(false, a, false, b, (short)0, c,
                                                false, false);
}

// Intra-wave LDS producer->consumer fence (DS ops are in-order per wave).
__device__ __forceinline__ void lds_fence() {
  asm volatile("s_wait_dscnt 0" ::: "memory");
}

// CDNA5 async global->LDS copy (ASYNCcnt-tracked), GVS addressing:
//   LDS[lds_addr] = MEM[saddr + voff], 16 bytes
__device__ __forceinline__ void async_load_b128(unsigned lds_addr, unsigned voff,
                                                const void* sbase) {
  asm volatile("global_load_async_to_lds_b128 %0, %1, %2"
               :: "v"(lds_addr), "v"(voff), "s"(sbase) : "memory");
}
__device__ __forceinline__ void wait_asynccnt0() {
  asm volatile("s_wait_asynccnt 0" ::: "memory");
}

// ---------------------------------------------------------------------------
// One-shot f32 -> f16 conversion (used for the 4 weight matrices)
// ---------------------------------------------------------------------------
__global__ __launch_bounds__(256) void cvt_f16_kernel(const float* __restrict__ src,
                                                      _Float16* __restrict__ dst) {
  int i = blockIdx.x * 256 + threadIdx.x;     // one float4 per thread
  float4 f = ((const float4*)src)[i];
  v4h h = {(_Float16)f.x, (_Float16)f.y, (_Float16)f.z, (_Float16)f.w};
  ((v4h*)dst)[i] = h;
}

// ---------------------------------------------------------------------------
// Y[m,n] = sum_k X[m,k]*Wh[n,k] + bias[n]
// W k-slabs (64 cols x 32 k, 4 KB f16) double-buffered in LDS via async copy;
// shared by all 4 waves (they compute the same column strip).
// PROJ=1: X fp32, Y f16 in [B,H,S,DH] head layout. PROJ=0: X f16, Y fp32 flat.
// ---------------------------------------------------------------------------
template <int PROJ>
__global__ __launch_bounds__(128) void gemm_kernel(
    const void* __restrict__ Xv, const _Float16* __restrict__ Wh,
    const float* __restrict__ bias, void* __restrict__ Yv) {
  __shared__ uint4 wtile_u[2][256];           // 2 x 4 KB
  _Float16* wl = (_Float16*)wtile_u;
  const unsigned wlds0 = (unsigned)(uintptr_t)wl;

  const int tid  = threadIdx.x;
  const int lane = tid & 31;
  const int wave = tid >> 5;
  const int hg   = lane >> 4;                 // 0: lanes 0-15, 1: lanes 16-31
  const int ln   = lane & 15;
  const int row0 = blockIdx.y * 64 + wave * 16;
  const int col0 = blockIdx.x * 64;

  // staging geometry: 256 x 16B per slab; col = idx>>2, 16B part = idx&3
  const int s0 = tid, s1 = tid + 128;
  const unsigned svo0 = (unsigned)((col0 + (s0 >> 2)) * (DM * 2) + (s0 & 3) * 16);
  const unsigned svo1 = (unsigned)((col0 + (s1 >> 2)) * (DM * 2) + (s1 & 3) * 16);
  const unsigned slo0 = (unsigned)((s0 >> 2) * 64 + (s0 & 3) * 16);
  const unsigned slo1 = (unsigned)((s1 >> 2) * 64 + (s1 & 3) * 16);

  {                                           // preload slab k0=0 into buf 0
    const char* wg = (const char*)Wh;
    async_load_b128(wlds0 + slo0, svo0, wg);
    async_load_b128(wlds0 + slo1, svo1, wg);
    wait_asynccnt0();
  }
  __syncthreads();

  v8f acc[4] = {};
  const float*    xrowf = (const float*)Xv + (size_t)(row0 + ln) * DM;
  const _Float16* xrowh = (const _Float16*)Xv + (size_t)(row0 + ln) * DM;

  int buf = 0;
  for (int k0 = 0; k0 < DM; k0 += 32) {
    if (k0 + 32 < DM) {                       // issue next slab into buf^1
      const char* wg = (const char*)Wh + (size_t)(k0 + 32) * 2;
      unsigned dst = wlds0 + (unsigned)(buf ^ 1) * 4096u;
      async_load_b128(dst + slo0, svo0, wg);
      async_load_b128(dst + slo1, svo1, wg);
    }
    v16h a;                                   // A 16x32: row = ln, K per-layout
#pragma unroll
    for (int e = 0; e < 16; ++e) {
      int kk = k0 + 16 * hg + e;
      a[e] = PROJ ? (_Float16)xrowf[kk] : xrowh[kk];
    }
    const _Float16* wb = wl + buf * 2048;
#pragma unroll
    for (int nt = 0; nt < 4; ++nt) {          // B column = lane, from LDS slab
      const _Float16* wr = wb + (nt * 16 + ln) * 32 + 16 * hg;
      v16h bm;
#pragma unroll
      for (int e = 0; e < 16; ++e) bm[e] = wr[e];
      acc[nt] = wmma_f16(a, bm, acc[nt]);
    }
    wait_asynccnt0();                         // next slab landed
    __syncthreads();
    buf ^= 1;
  }

#pragma unroll
  for (int nt = 0; nt < 4; ++nt) {
    int n = col0 + nt * 16 + ln;
    float bv = bias[n];
#pragma unroll
    for (int i = 0; i < 8; ++i) {
      int m = row0 + i + 8 * hg;              // C row = VGPR idx + 8*hg
      float o = acc[nt][i] + bv;
      if (PROJ) {
        int h = n >> 6, dh = n & 63, bb = m >> 11, s = m & 2047;
        ((_Float16*)Yv)[(((size_t)bb * NH + h) * SEQ + s) * DH + dh] = (_Float16)o;
      } else {
        ((float*)Yv)[(size_t)m * DM + n] = o;
      }
    }
  }
}

// ---------------------------------------------------------------------------
// Flash-style attention per (b, h). Block = 4 waves; wave owns 16 query rows.
// K/V chunks (32 keys) double-buffered in LDS via CDNA5 async global->LDS,
// overlapped with WMMA compute. Masked scores -> 1e-12 (reference quirk).
// ---------------------------------------------------------------------------
__global__ __launch_bounds__(128) void attn_kernel(
    const _Float16* __restrict__ Qh, const _Float16* __restrict__ Kh,
    const _Float16* __restrict__ Vh, const int* __restrict__ masks,
    _Float16* __restrict__ attn_out) {
  __shared__ uint4 kbuf_u[2][256];            // 2 x (32 keys x 64 f16) = 2x4 KB
  __shared__ uint4 vbuf_u[2][256];
  __shared__ _Float16 p_lds[4][512];          // per-wave 16x32 P tile
  _Float16* kfull = (_Float16*)kbuf_u;
  _Float16* vfull = (_Float16*)vbuf_u;
  const unsigned klds0 = (unsigned)(uintptr_t)kfull;
  const unsigned vlds0 = (unsigned)(uintptr_t)vfull;

  const int tid  = threadIdx.x;
  const int lane = tid & 31, wave = tid >> 5;
  const int hg   = lane >> 4, ln = lane & 15;
  const int b    = blockIdx.z, h = blockIdx.y;
  const int q0   = blockIdx.x * 64 + wave * 16;
  const size_t headbase = ((size_t)b * NH + h) * SEQ;

  const unsigned boff0 = (unsigned)tid * 16u;         // staging: 2x16B/thread
  const unsigned boff1 = (unsigned)(tid + 128) * 16u;

  {                                           // preload chunk 0 into buf 0
    const char* kg = (const char*)(Kh + headbase * DH);
    const char* vg = (const char*)(Vh + headbase * DH);
    async_load_b128(klds0 + boff0, boff0, kg);
    async_load_b128(klds0 + boff1, boff1, kg);
    async_load_b128(vlds0 + boff0, boff0, vg);
    async_load_b128(vlds0 + boff1, boff1, vg);
    wait_asynccnt0();
  }
  __syncthreads();

  // Q A-operands (dh 0..31 and 32..63), pre-scaled by 1/sqrt(64)
  v16h qa0, qa1;
  {
    const _Float16* qrow = Qh + (headbase + (size_t)(q0 + ln)) * DH;
#pragma unroll
    for (int e = 0; e < 16; ++e) {
      int kk = 16 * hg + e;
      qa0[e] = (_Float16)((float)qrow[kk] * 0.125f);
      qa1[e] = (_Float16)((float)qrow[kk + 32] * 0.125f);
    }
  }

  v8f acc[4] = {};
  float mrow[8], lrow[8];
#pragma unroll
  for (int i = 0; i < 8; ++i) { mrow[i] = -1e30f; lrow[i] = 0.0f; }

  const int* mbase = masks + (size_t)b * SEQ * SEQ;

  int buf = 0;
  for (int t0 = 0; t0 < SEQ; t0 += 32) {
    if (t0 + 32 < SEQ) {                      // issue next K/V chunk -> buf^1
      const char* kg = (const char*)(Kh + (headbase + t0 + 32) * DH);
      const char* vg = (const char*)(Vh + (headbase + t0 + 32) * DH);
      unsigned kd = klds0 + (unsigned)(buf ^ 1) * 4096u;
      unsigned vd = vlds0 + (unsigned)(buf ^ 1) * 4096u;
      async_load_b128(kd + boff0, boff0, kg);
      async_load_b128(kd + boff1, boff1, kg);
      async_load_b128(vd + boff0, boff0, vg);
      async_load_b128(vd + boff1, boff1, vg);
      // prefetch next chunk of the mask stream (the HBM-bound input)
      __builtin_prefetch(mbase + (size_t)(q0 + ln) * SEQ + t0 + 32, 0, 0);
    }
    const _Float16* k_lds = kfull + buf * 2048;
    const _Float16* v_lds = vfull + buf * 2048;

    // ---- scores for two 16-key tiles, masked ----
    float pv[2][8];
#pragma unroll
    for (int kt = 0; kt < 2; ++kt) {
      v8f s = {};
      {
        v16h b0, b1;
        const _Float16* kc = k_lds + (size_t)(kt * 16 + ln) * DH + 16 * hg;
#pragma unroll
        for (int e = 0; e < 16; ++e) { b0[e] = kc[e]; b1[e] = kc[e + 32]; }
        s = wmma_f16(qa0, b0, s);
        s = wmma_f16(qa1, b1, s);
      }
      const int key = t0 + kt * 16 + ln;
#pragma unroll
      for (int i = 0; i < 8; ++i) {
        int row = q0 + i + 8 * hg;
        int mv = mbase[(size_t)row * SEQ + key];
        pv[kt][i] = mv ? s[i] : 1e-12f;       // reference quirk
      }
    }

    // ---- online softmax update over the 32-key chunk ----
#pragma unroll
    for (int i = 0; i < 8; ++i) {
      float v = fmaxf(pv[0][i], pv[1][i]);
#pragma unroll
      for (int off = 1; off < 16; off <<= 1)
        v = fmaxf(v, __shfl_xor(v, off, 32)); // row-max over 16-lane group
      float mnew = fmaxf(mrow[i], v);
      float corr = __expf(mrow[i] - mnew);
      mrow[i] = mnew;
      float p0 = __expf(pv[0][i] - mnew);
      float p1 = __expf(pv[1][i] - mnew);
      float ps = p0 + p1;
#pragma unroll
      for (int off = 1; off < 16; off <<= 1)
        ps += __shfl_xor(ps, off, 32);        // row-sum over 16-lane group
      lrow[i] = lrow[i] * corr + ps;
#pragma unroll
      for (int nt = 0; nt < 4; ++nt) acc[nt][i] *= corr;
      int prow = (i + 8 * hg) * 32;
      p_lds[wave][prow + ln]      = (_Float16)p0;
      p_lds[wave][prow + 16 + ln] = (_Float16)p1;
    }
    lds_fence();                              // C-layout -> A-layout via LDS

    // ---- P(16x32) @ V(32x64) ----
    v16h pa;
    {
      const _Float16* pr = &p_lds[wave][ln * 32];
#pragma unroll
      for (int e = 0; e < 16; ++e) {
        int kk = 16 * hg + e;
        pa[e] = pr[kk];
      }
    }
#pragma unroll
    for (int nt = 0; nt < 4; ++nt) {
      v16h vb;
      const _Float16* vc = v_lds + (size_t)(16 * hg) * DH + nt * 16 + ln;
#pragma unroll
      for (int e = 0; e < 16; ++e) vb[e] = vc[(size_t)e * DH];
      acc[nt] = wmma_f16(pa, vb, acc[nt]);
    }

    wait_asynccnt0();                         // t0+32 chunk landed
    __syncthreads();
    buf ^= 1;
  }

  // ---- normalize + store f16 into [B,S,DM] (heads concatenated) ----
#pragma unroll
  for (int nt = 0; nt < 4; ++nt) {
    int dh = nt * 16 + ln;
#pragma unroll
    for (int i = 0; i < 8; ++i) {
      int srow = q0 + i + 8 * hg;
      float o = acc[nt][i] / lrow[i];
      attn_out[((size_t)b * SEQ + srow) * DM + h * DH + dh] = (_Float16)o;
    }
  }
}

// ---------------------------------------------------------------------------
extern "C" void kernel_launch(void* const* d_in, const int* in_sizes, int n_in,
                              void* d_out, int out_size, void* d_ws, size_t ws_size,
                              hipStream_t stream) {
  const float* q     = (const float*)d_in[0];
  const float* k     = (const float*)d_in[1];
  const float* v     = (const float*)d_in[2];
  const int*   masks = (const int*)d_in[3];
  const float* WQ_w  = (const float*)d_in[4];
  const float* WQ_b  = (const float*)d_in[5];
  const float* WK_w  = (const float*)d_in[6];
  const float* WK_b  = (const float*)d_in[7];
  const float* WV_w  = (const float*)d_in[8];
  const float* WV_b  = (const float*)d_in[9];
  const float* WO_w  = (const float*)d_in[10];
  const float* WO_b  = (const float*)d_in[11];
  float* out = (float*)d_out;

  // workspace: Qh|Kh|Vh ([B,H,S,DH] f16) | attn ([B*S,DM] f16) | 4x Wh f16
  const size_t headElems = (size_t)BATCH * NH * SEQ * DH;   // 3,145,728
  const size_t wElems    = (size_t)DM * DM;                 //   589,824
  _Float16* Qh   = (_Float16*)d_ws;
  _Float16* Kh   = Qh + headElems;
  _Float16* Vh   = Kh + headElems;
  _Float16* attn = Vh + headElems;
  _Float16* WhQ  = attn + headElems;
  _Float16* WhK  = WhQ + wElems;
  _Float16* WhV  = WhK + wElems;
  _Float16* WhO  = WhV + wElems;

  dim3 blk(128);
  dim3 cgrid(wElems / 4 / 256);               // 576 blocks, float4 per thread
  cvt_f16_kernel<<<cgrid, 256, 0, stream>>>(WQ_w, WhQ);
  cvt_f16_kernel<<<cgrid, 256, 0, stream>>>(WK_w, WhK);
  cvt_f16_kernel<<<cgrid, 256, 0, stream>>>(WV_w, WhV);
  cvt_f16_kernel<<<cgrid, 256, 0, stream>>>(WO_w, WhO);

  dim3 ggrid(DM / 64, (BATCH * SEQ) / 64);
  gemm_kernel<1><<<ggrid, blk, 0, stream>>>(q, WhQ, WQ_b, Qh);
  gemm_kernel<1><<<ggrid, blk, 0, stream>>>(k, WhK, WK_b, Kh);
  gemm_kernel<1><<<ggrid, blk, 0, stream>>>(v, WhV, WV_b, Vh);
  attn_kernel<<<dim3(SEQ / 64, NH, BATCH), blk, 0, stream>>>(Qh, Kh, Vh, masks, attn);
  gemm_kernel<0><<<ggrid, blk, 0, stream>>>(attn, WhO, WO_b, out);
}
